// residual_graph_attn_block_33938831573431
// MI455X (gfx1250) — compile-verified, matrix operand order checked by reading, not run on hardware
//
#include <hip/hip_runtime.h>
#include <cmath>

#define NODES 50000
#define EDGES 800000
#define FIN   64
#define HC    128

typedef __attribute__((ext_vector_type(16))) __bf16 v16bf;
typedef __attribute__((ext_vector_type(8)))  float  v8f;

// ---------------------------------------------------------------- WMMA helpers
__device__ __forceinline__ v8f wmma_bf16(v16bf a, v16bf b, v8f c) {
  // D = A(16x32 bf16) x B(32x16 bf16) + C(16x16 f32)
  return __builtin_amdgcn_wmma_f32_16x16x32_bf16(false, a, false, b, (short)0, c,
                                                 false, false);
}

// A-matrix 16x32 bf16 layout (wave32): lane = 16*hi + ln holds row=ln.
// low lanes: elems 0..7 -> K=k0+0..7,  elems 8..15 -> K=k0+16..23
// hi  lanes: elems 0..7 -> K=k0+8..15, elems 8..15 -> K=k0+24..31
__device__ __forceinline__ v16bf load_A_f32(const float* __restrict__ p, int ld,
                                            int row, int hi, int k0) {
  const float* r = p + (size_t)row * ld + k0 + (hi ? 8 : 0);
  v16bf a;
#pragma unroll
  for (int i = 0; i < 8; ++i) a[i] = (__bf16)r[i];
#pragma unroll
  for (int i = 0; i < 8; ++i) a[8 + i] = (__bf16)r[16 + i];
  return a;
}

// A chunk from agg (+gat_bias) rows
__device__ __forceinline__ v16bf load_A_gat(const float* __restrict__ agg,
                                            const float* __restrict__ gb,
                                            int row, int hi, int c0) {
  int off = hi ? 8 : 0;
  const float* r = agg + (size_t)row * HC + c0 + off;
  const float* g = gb + c0 + off;
  v16bf a;
#pragma unroll
  for (int i = 0; i < 8; ++i) a[i] = (__bf16)(r[i] + g[i]);
#pragma unroll
  for (int i = 0; i < 8; ++i) a[8 + i] = (__bf16)(r[16 + i] + g[16 + i]);
  return a;
}

// A from edge features [E,16], K padded 16->32 with zeros
__device__ __forceinline__ v16bf load_A_edge(const float* __restrict__ EF,
                                             int row, int hi) {
  const float* r = EF + (size_t)row * 16 + (hi ? 8 : 0);
  v16bf a;
#pragma unroll
  for (int i = 0; i < 8; ++i) a[i] = (__bf16)r[i];
#pragma unroll
  for (int i = 0; i < 8; ++i) a[8 + i] = (__bf16)0.0f;
  return a;
}

// Pre-packed B operand: one v16bf (32 contiguous bytes) per (chunk,col,hi)
__device__ __forceinline__ v16bf load_B_pk(const v16bf* __restrict__ pw, int chunk,
                                           int col, int hi) {
  return pw[(((size_t)chunk * HC + col) << 1) | hi];
}

// sum across the 16 lanes of a half-wave (matches WMMA C-layout columns)
__device__ __forceinline__ float half16_reduce(float v) {
  v += __shfl_xor(v, 1, 32);
  v += __shfl_xor(v, 2, 32);
  v += __shfl_xor(v, 4, 32);
  v += __shfl_xor(v, 8, 32);
  return v;
}

__device__ __forceinline__ void atomicMaxFloat(float* addr, float val) {
  if (val >= 0.0f) atomicMax((int*)addr, __float_as_int(val));
  else             atomicMin((unsigned int*)addr, (unsigned int)__float_as_int(val));
}

// ---------------------------------------------------------------- kernels
__global__ void k_init(float* __restrict__ mx, float* __restrict__ den,
                       float* __restrict__ agg) {
  int i = blockIdx.x * blockDim.x + threadIdx.x;
  if (i < NODES * 4) { mx[i] = -__builtin_inff(); den[i] = 0.0f; }
  if (i < NODES * HC) agg[i] = 0.0f;
}

// Pack a [kvalid x 128] f32 weight matrix into WMMA B-layout bf16 chunks of K=32.
// One thread per (chunk, col, hi) -> writes 16 bf16 (32 bytes).
__global__ void k_pack_w(const float* __restrict__ W, int kvalid, int chunks,
                         v16bf* __restrict__ out) {
  int i = blockIdx.x * blockDim.x + threadIdx.x;
  int total = chunks * HC * 2;
  if (i >= total) return;
  int hi = i & 1;
  int col = (i >> 1) & (HC - 1);
  int chunk = i >> 8;
  int kb = chunk * 32 + (hi ? 8 : 0);
  v16bf v;
#pragma unroll
  for (int j = 0; j < 8; ++j) {
    int k = kb + j;
    v[j] = (__bf16)((k < kvalid) ? W[(size_t)k * HC + col] : 0.0f);
  }
#pragma unroll
  for (int j = 0; j < 8; ++j) {
    int k = kb + 16 + j;
    v[8 + j] = (__bf16)((k < kvalid) ? W[(size_t)k * HC + col] : 0.0f);
  }
  out[i] = v;
}

__device__ __forceinline__ void node_gemm_store(v16bf a0, v16bf a1,
    const v16bf* __restrict__ pkW, const float* __restrict__ bias,
    float* __restrict__ outbuf, int nb, int ln, int hi) {
#pragma unroll
  for (int t = 0; t < 8; ++t) {
    int col = t * 16 + ln;
    v8f c = {};
    c = wmma_bf16(a0, load_B_pk(pkW, 0, col, hi), c);
    c = wmma_bf16(a1, load_B_pk(pkW, 1, col, hi), c);
    float bv = bias[col];
#pragma unroll
    for (int r = 0; r < 8; ++r)
      outbuf[(size_t)(nb + r + 8 * hi) * HC + col] = c[r] + bv;
  }
}

__global__ void k_node_xform(const float* __restrict__ X,
                             const v16bf* __restrict__ pkWl, const float* __restrict__ bl,
                             const v16bf* __restrict__ pkWr, const float* __restrict__ br,
                             const v16bf* __restrict__ pkWres, const float* __restrict__ bres,
                             float* __restrict__ xl, float* __restrict__ xr,
                             float* __restrict__ resp) {
  int gw = (blockIdx.x * blockDim.x + threadIdx.x) >> 5;
  if (gw >= NODES / 16) return;
  int lane = threadIdx.x & 31;
  int ln = lane & 15, hi = lane >> 4;
  int nb = gw * 16;
  int row = nb + ln;
  v16bf a0 = load_A_f32(X, FIN, row, hi, 0);
  v16bf a1 = load_A_f32(X, FIN, row, hi, 32);
  node_gemm_store(a0, a1, pkWl, bl, xl, nb, ln, hi);
  node_gemm_store(a0, a1, pkWr, br, xr, nb, ln, hi);
  node_gemm_store(a0, a1, pkWres, bres, resp, nb, ln, hi);
}

__global__ void k_edge_logits(const float* __restrict__ EF,
                              const v16bf* __restrict__ pkWe,
                              const int* __restrict__ src, const int* __restrict__ dst,
                              const float* __restrict__ xl, const float* __restrict__ xr,
                              const float* __restrict__ att,
                              float* __restrict__ logits, float* __restrict__ mx) {
  int gw = (blockIdx.x * blockDim.x + threadIdx.x) >> 5;
  int lane = threadIdx.x & 31;
  int ln = lane & 15, hi = lane >> 4;
  int ebase = gw * 16;

  // indices first, then prefetch gather rows so WMMA work hides the latency
  int srcs[8], dsts[8];
#pragma unroll
  for (int r = 0; r < 8; ++r) {
    int e = ebase + r + 8 * hi;
    srcs[r] = src[e];
    dsts[r] = dst[e];
  }
#pragma unroll
  for (int r = 0; r < 8; ++r) {
    __builtin_prefetch(&xl[(size_t)srcs[r] * HC + ln * 8], 0, 1);
    __builtin_prefetch(&xr[(size_t)dsts[r] * HC + ln * 8], 0, 1);
  }

  // e = EF @ W_e for this 16-edge tile, kept in registers (never spilled to HBM)
  v16bf a = load_A_edge(EF, ebase + ln, hi);
  v8f acc[8];
#pragma unroll
  for (int t = 0; t < 8; ++t) {
    v8f c = {};
    acc[t] = wmma_bf16(a, load_B_pk(pkWe, 0, t * 16 + ln, hi), c);
  }

  float attv[8];
#pragma unroll
  for (int t = 0; t < 8; ++t) attv[t] = att[t * 16 + ln];

#pragma unroll
  for (int r = 0; r < 8; ++r) {
    float p[4] = {0.f, 0.f, 0.f, 0.f};
#pragma unroll
    for (int t = 0; t < 8; ++t) {
      int n = t * 16 + ln;
      float m = acc[t][r] + xl[(size_t)srcs[r] * HC + n] + xr[(size_t)dsts[r] * HC + n];
      float f = (m >= 0.0f) ? m : 0.2f * m;
      p[t >> 1] += f * attv[t];
    }
#pragma unroll
    for (int h = 0; h < 4; ++h) p[h] = half16_reduce(p[h]);
    if (ln == 0) {
      int e = ebase + r + 8 * hi;
#pragma unroll
      for (int h = 0; h < 4; ++h) {
        logits[(size_t)e * 4 + h] = p[h];
        atomicMaxFloat(&mx[(size_t)dsts[r] * 4 + h], p[h]);
      }
    }
  }
}

__global__ void k_edge_exp(const int* __restrict__ dst, const float* __restrict__ mx,
                           float* __restrict__ lgt, float* __restrict__ den) {
  int i = blockIdx.x * blockDim.x + threadIdx.x;
  if (i >= EDGES * 4) return;
  int e = i >> 2, h = i & 3;
  int d = dst[e];
  float ex = __expf(lgt[i] - mx[(size_t)d * 4 + h]);
  lgt[i] = ex;
  atomicAdd(&den[(size_t)d * 4 + h], ex);
}

__global__ void k_aggregate(const int* __restrict__ src, const int* __restrict__ dst,
                            const float* __restrict__ ex, const float* __restrict__ den,
                            const float* __restrict__ xl, float* __restrict__ agg) {
  int t = blockIdx.x * blockDim.x + threadIdx.x;
  int e = t >> 5;
  if (e >= EDGES) return;
  int lane = t & 31;
  int c = lane * 4;
  int h = c >> 5;
  int s = src[e], d = dst[e];
  float alpha = ex[(size_t)e * 4 + h] / den[(size_t)d * 4 + h];
  float4 xv = *(const float4*)&xl[(size_t)s * HC + c];
  atomicAdd(&agg[(size_t)d * HC + c + 0], alpha * xv.x);
  atomicAdd(&agg[(size_t)d * HC + c + 1], alpha * xv.y);
  atomicAdd(&agg[(size_t)d * HC + c + 2], alpha * xv.z);
  atomicAdd(&agg[(size_t)d * HC + c + 3], alpha * xv.w);
}

__global__ void k_identity(const float* __restrict__ resp, const float* __restrict__ g,
                           const float* __restrict__ b, float* __restrict__ out) {
  int gw = (blockIdx.x * blockDim.x + threadIdx.x) >> 5;  // one wave per node row
  int lane = threadIdx.x & 31;
  if (gw >= NODES) return;
  int c = lane * 4;
  float4 v = *(const float4*)(resp + (size_t)gw * HC + c);
  float s1 = v.x + v.y + v.z + v.w;
  float s2 = v.x * v.x + v.y * v.y + v.z * v.z + v.w * v.w;
#pragma unroll
  for (int m = 1; m < 32; m <<= 1) {
    s1 += __shfl_xor(s1, m, 32);
    s2 += __shfl_xor(s2, m, 32);
  }
  float mu = s1 * (1.0f / HC);
  float var = s2 * (1.0f / HC) - mu * mu;
  float rs = rsqrtf(var + 1e-5f);
  float4 o;
  o.x = (v.x - mu) * rs * g[c + 0] + b[c + 0];
  o.y = (v.y - mu) * rs * g[c + 1] + b[c + 1];
  o.z = (v.z - mu) * rs * g[c + 2] + b[c + 2];
  o.w = (v.w - mu) * rs * g[c + 3] + b[c + 3];
  *(float4*)(out + (size_t)gw * HC + c) = o;
}

__global__ void k_mlp(const float* __restrict__ X, const float* __restrict__ agg,
                      const float* __restrict__ gb,
                      const v16bf* __restrict__ pkW0, const float* __restrict__ b0,
                      const float* __restrict__ g0, const float* __restrict__ be0,
                      const v16bf* __restrict__ pkW1, const float* __restrict__ b1,
                      const float* __restrict__ g1, const float* __restrict__ be1,
                      float* __restrict__ out) {
  __shared__ float lds[8 * 16 * HC];  // 64 KB: one 16x128 f32 tile per wave
  int w = threadIdx.x >> 5;
  int gw = (blockIdx.x * blockDim.x + threadIdx.x) >> 5;
  int lane = threadIdx.x & 31;
  int ln = lane & 15, hi = lane >> 4;
  bool valid = gw < (NODES / 16);
  int tile = valid ? gw : 0;  // clamp so all waves reach __syncthreads
  int nb = tile * 16;
  int row = nb + ln;

  // ---- GEMM0: [16 x 192] @ W0[192 x 128], A = concat(X, agg + gat_bias)
  v16bf a0 = load_A_f32(X, FIN, row, hi, 0);
  v16bf a1 = load_A_f32(X, FIN, row, hi, 32);
  v16bf a2 = load_A_gat(agg, gb, row, hi, 0);
  v16bf a3 = load_A_gat(agg, gb, row, hi, 32);
  v16bf a4 = load_A_gat(agg, gb, row, hi, 64);
  v16bf a5 = load_A_gat(agg, gb, row, hi, 96);

  v8f acc[8];
#pragma unroll
  for (int t = 0; t < 8; ++t) {
    int col = t * 16 + ln;
    v8f c = {};
    c = wmma_bf16(a0, load_B_pk(pkW0, 0, col, hi), c);
    c = wmma_bf16(a1, load_B_pk(pkW0, 1, col, hi), c);
    c = wmma_bf16(a2, load_B_pk(pkW0, 2, col, hi), c);
    c = wmma_bf16(a3, load_B_pk(pkW0, 3, col, hi), c);
    c = wmma_bf16(a4, load_B_pk(pkW0, 4, col, hi), c);
    c = wmma_bf16(a5, load_B_pk(pkW0, 5, col, hi), c);
    float bv = b0[col];
#pragma unroll
    for (int r = 0; r < 8; ++r) c[r] += bv;
    acc[t] = c;
  }

  // ---- LayerNorm0 + ReLU (row stats via half-wave shuffle reduce)
  float mu[8], rs[8];
#pragma unroll
  for (int r = 0; r < 8; ++r) {
    float s1 = 0.f, s2 = 0.f;
#pragma unroll
    for (int t = 0; t < 8; ++t) { float v = acc[t][r]; s1 += v; s2 += v * v; }
    s1 = half16_reduce(s1);
    s2 = half16_reduce(s2);
    float m = s1 * (1.0f / HC);
    mu[r] = m;
    rs[r] = rsqrtf(s2 * (1.0f / HC) - m * m + 1e-5f);
  }
  float* my = &lds[w * 16 * HC];
#pragma unroll
  for (int t = 0; t < 8; ++t) {
    int n = t * 16 + ln;
    float gg = g0[n], bb = be0[n];
#pragma unroll
    for (int r = 0; r < 8; ++r) {
      float v = (acc[t][r] - mu[r]) * rs[r] * gg + bb;
      my[(r + 8 * hi) * HC + n] = fmaxf(v, 0.0f);
    }
  }
  __syncthreads();

  // ---- GEMM1: [16 x 128] @ W1[128 x 128]
  v16bf h0 = load_A_f32(my, HC, ln, hi, 0);
  v16bf h1 = load_A_f32(my, HC, ln, hi, 32);
  v16bf h2 = load_A_f32(my, HC, ln, hi, 64);
  v16bf h3 = load_A_f32(my, HC, ln, hi, 96);
#pragma unroll
  for (int t = 0; t < 8; ++t) {
    int col = t * 16 + ln;
    v8f c = {};
    c = wmma_bf16(h0, load_B_pk(pkW1, 0, col, hi), c);
    c = wmma_bf16(h1, load_B_pk(pkW1, 1, col, hi), c);
    c = wmma_bf16(h2, load_B_pk(pkW1, 2, col, hi), c);
    c = wmma_bf16(h3, load_B_pk(pkW1, 3, col, hi), c);
    float bv = b1[col];
#pragma unroll
    for (int r = 0; r < 8; ++r) c[r] += bv;
    acc[t] = c;
  }

  // ---- LayerNorm1 + ReLU, accumulate onto identity in d_out
#pragma unroll
  for (int r = 0; r < 8; ++r) {
    float s1 = 0.f, s2 = 0.f;
#pragma unroll
    for (int t = 0; t < 8; ++t) { float v = acc[t][r]; s1 += v; s2 += v * v; }
    s1 = half16_reduce(s1);
    s2 = half16_reduce(s2);
    float m = s1 * (1.0f / HC);
    mu[r] = m;
    rs[r] = rsqrtf(s2 * (1.0f / HC) - m * m + 1e-5f);
  }
#pragma unroll
  for (int t = 0; t < 8; ++t) {
    int n = t * 16 + ln;
    float gg = g1[n], bb = be1[n];
#pragma unroll
    for (int r = 0; r < 8; ++r) {
      float v = (acc[t][r] - mu[r]) * rs[r] * gg + bb;
      v = fmaxf(v, 0.0f);
      if (valid) {
        float* o = &out[(size_t)(nb + r + 8 * hi) * HC + n];
        *o = *o + v;
      }
    }
  }
}

// ---------------------------------------------------------------- launch
extern "C" void kernel_launch(void* const* d_in, const int* in_sizes, int n_in,
                              void* d_out, int out_size, void* d_ws, size_t ws_size,
                              hipStream_t stream) {
  (void)in_sizes; (void)n_in; (void)out_size; (void)ws_size;
  const float* X    = (const float*)d_in[0];
  const float* EF   = (const float*)d_in[1];
  const int*   EI   = (const int*)d_in[2];
  const float* Wl   = (const float*)d_in[3];
  const float* bl   = (const float*)d_in[4];
  const float* Wr   = (const float*)d_in[5];
  const float* br   = (const float*)d_in[6];
  const float* We   = (const float*)d_in[7];
  const float* att  = (const float*)d_in[8];
  const float* gb   = (const float*)d_in[9];
  const float* Wres = (const float*)d_in[10];
  const float* bres = (const float*)d_in[11];
  const float* rg   = (const float*)d_in[12];
  const float* rb   = (const float*)d_in[13];
  const float* W0   = (const float*)d_in[14];
  const float* b0   = (const float*)d_in[15];
  const float* g0   = (const float*)d_in[16];
  const float* be0  = (const float*)d_in[17];
  const float* W1   = (const float*)d_in[18];
  const float* b1   = (const float*)d_in[19];
  const float* g1   = (const float*)d_in[20];
  const float* be1  = (const float*)d_in[21];
  float* out = (float*)d_out;

  float* ws   = (float*)d_ws;
  float* xl   = ws;
  float* xr   = xl + (size_t)NODES * HC;
  float* resp = xr + (size_t)NODES * HC;
  float* agg  = resp + (size_t)NODES * HC;
  float* lgt  = agg + (size_t)NODES * HC;
  float* mx   = lgt + (size_t)EDGES * 4;
  float* den  = mx + (size_t)NODES * 4;
  // packed bf16 weights (32-byte aligned: all offsets are multiples of 8 floats)
  v16bf* pkWl   = (v16bf*)(den + (size_t)NODES * 4);
  v16bf* pkWr   = pkWl + 2 * HC * 2;   // 2 K-chunks
  v16bf* pkWres = pkWr + 2 * HC * 2;
  v16bf* pkWe   = pkWres + 2 * HC * 2; // 1 K-chunk (K=16 zero-padded)
  v16bf* pkW0   = pkWe + 1 * HC * 2;   // 6 K-chunks
  v16bf* pkW1   = pkW0 + 6 * HC * 2;   // 4 K-chunks

  const int* srcp = EI;
  const int* dstp = EI + EDGES;

  // one-shot weight packing (tiny)
  k_pack_w<<<dim3(2), dim3(256), 0, stream>>>(Wl, 64, 2, pkWl);
  k_pack_w<<<dim3(2), dim3(256), 0, stream>>>(Wr, 64, 2, pkWr);
  k_pack_w<<<dim3(2), dim3(256), 0, stream>>>(Wres, 64, 2, pkWres);
  k_pack_w<<<dim3(1), dim3(256), 0, stream>>>(We, 16, 1, pkWe);
  k_pack_w<<<dim3(6), dim3(256), 0, stream>>>(W0, 192, 6, pkW0);
  k_pack_w<<<dim3(4), dim3(256), 0, stream>>>(W1, 128, 4, pkW1);

  k_init<<<dim3((NODES * HC) / 256), dim3(256), 0, stream>>>(mx, den, agg);
  k_node_xform<<<dim3((NODES / 16 + 7) / 8), dim3(256), 0, stream>>>(
      X, pkWl, bl, pkWr, br, pkWres, bres, xl, xr, resp);
  k_edge_logits<<<dim3(EDGES / 16 / 8), dim3(256), 0, stream>>>(
      EF, pkWe, srcp, dstp, xl, xr, att, lgt, mx);
  k_edge_exp<<<dim3((EDGES * 4) / 256), dim3(256), 0, stream>>>(dstp, mx, lgt, den);
  k_aggregate<<<dim3((EDGES * 32) / 256), dim3(256), 0, stream>>>(
      srcp, dstp, lgt, den, xl, agg);
  k_identity<<<dim3(NODES / 8), dim3(256), 0, stream>>>(resp, rg, rb, out);
  k_mlp<<<dim3((NODES / 16 + 7) / 8), dim3(256), 0, stream>>>(
      X, agg, gb, pkW0, b0, g0, be0, pkW1, b1, g1, be1, out);
}
